// KManifoldAEClusterModel_6493990552285
// MI455X (gfx1250) — matrix operand order repaired and checked by reading, not compiled, not run on hardware
//
#include <hip/hip_runtime.h>
#include <hip/hip_bf16.h>
#include <stdint.h>

// ---------------------------------------------------------------------------
// KManifoldAE cluster objective, MI455X (gfx1250, wave32, WMMA).
//
//   out = mean_b min_k ( ||x_b||^2 - 2*w_k.z_k + z_k^T G_k z_k + LAMB*reg[k] )
//   z_k = V_k x_b, w_k = U_k^T x_b, G_k = U_k^T U_k
//
// One bf16 GEMM  Y = X(16384x512) * P^T(512x320)  at the HBM roofline
// (~33MB -> ~1.4us @ 23.3TB/s), fp32 accumulate via v_wmma_f32_16x16x32_bf16.
// ---------------------------------------------------------------------------

typedef __attribute__((ext_vector_type(16))) __bf16 v16bf;
typedef __attribute__((ext_vector_type(8)))  float  v8f;

#define K_CL   16
#define D_AMBI 512
#define D_LAT  10
#define BATCH  16384
#define LAMB_W 0.01f

#define NCOL       320            // 2 * K_CL * D_LAT columns of P
#define WG_THREADS 640            // 20 waves: wave w owns n-tile w (16 cols)
#define NWG        512
#define BT_PER     (BATCH / 16 / NWG)   // 2 batch tiles per WG

#define XPITCH 1040               // bytes per LDS row of bf16 X tile (520 bf16, padded)
#define YPITCH 324                // floats per LDS row of Y (padded vs 320)

// workspace layout (bytes)
#define WS_P    0                         // bf16 P[320][512]      = 327680 B
#define WS_G    (NCOL * D_AMBI * 2)       // f32  G[16][100]       = 6400 B
#define WS_REG  (WS_G + K_CL * 100 * 4)   // f32  reg[16]          = 64 B
#define WS_PART (WS_REG + 64)             // f32  partials[NWG]    = 2048 B

// --- P packing: rows 0..159 = Vs rows (flat copy); rows 160..319 = Us columns.
__global__ void kmae_prep_P(const float* __restrict__ Us,
                            const float* __restrict__ Vs,
                            __bf16* __restrict__ P) {
  int r = blockIdx.x;
  for (int d = threadIdx.x; d < D_AMBI; d += blockDim.x) {
    float v;
    if (r < K_CL * D_LAT) {
      v = Vs[r * D_AMBI + d];                       // Vs[k][j][d], r = k*10+j
    } else {
      int rr = r - K_CL * D_LAT;
      int k = rr / D_LAT, j = rr % D_LAT;
      v = Us[(k * D_AMBI + d) * D_LAT + j];         // Us[k][d][j]
    }
    P[r * D_AMBI + d] = (__bf16)v;
  }
}

// --- G_k = U_k^T U_k (fp32), reg[k] = 0.5*(||U_k||_F^2 + ||V_k||_F^2)
__global__ void kmae_prep_G(const float* __restrict__ Us,
                            const float* __restrict__ Vs,
                            float* __restrict__ G,
                            float* __restrict__ reg) {
  __shared__ float Gs[100];
  __shared__ float red[128];
  int k = blockIdx.x, t = threadIdx.x;
  if (t < 100) {
    int i = t / 10, j = t % 10;
    float s = 0.f;
    for (int d = 0; d < D_AMBI; ++d)
      s = fmaf(Us[(k * D_AMBI + d) * D_LAT + i],
               Us[(k * D_AMBI + d) * D_LAT + j], s);
    G[k * 100 + t] = s;
    Gs[t] = s;
  }
  float pv = 0.f;
  for (int idx = t; idx < D_LAT * D_AMBI; idx += 128) {
    float v = Vs[k * D_LAT * D_AMBI + idx];
    pv = fmaf(v, v, pv);
  }
  red[t] = pv;
  __syncthreads();
  if (t == 0) {
    float sv = 0.f;
    for (int i = 0; i < 128; ++i) sv += red[i];
    float su = 0.f;
    for (int i = 0; i < 10; ++i) su += Gs[i * 10 + i];  // trace(G) = ||U||_F^2
    reg[k] = 0.5f * (su + sv);
  }
}

// --- Main: GEMM tiles + fused epilogue.
__global__ __launch_bounds__(WG_THREADS) void kmae_main(
    const float* __restrict__ x, const __bf16* __restrict__ P,
    const float* __restrict__ G, const float* __restrict__ regw,
    float* __restrict__ partials) {
  __shared__ __align__(16) unsigned char xt[16 * XPITCH]; // bf16 X tile
  __shared__ __align__(16) float Yf[16 * YPITCH];         // f32 Y tile (16x320, padded)
  __shared__ float Gl[K_CL * 100];
  __shared__ float regl[K_CL];
  __shared__ float xx[16];
  __shared__ float minv[16];

  const int tid  = threadIdx.x;
  const int lane = tid & 31;
  const int w    = tid >> 5;                 // wave id 0..19 == n-tile id
  const int wg   = blockIdx.x;

  for (int i = tid; i < K_CL * 100; i += WG_THREADS) Gl[i] = G[i];
  if (tid < K_CL) regl[tid] = regw[tid];

  // Resident B fragments (32x16 bf16, 16 k-steps): contiguous K order, same as A —
  // lane 0..15: N=lane, K=0..15; lane 16..31: N=lane-16, K=16..31.
  v16bf Bf[16];
  {
    int r = w * 16 + (lane & 15);            // row of P == output column
    int h = (lane < 16) ? 0 : 1;             // K-half
    const unsigned char* pb =
        (const unsigned char*)P + (size_t)r * (D_AMBI * 2) + h * 32;
#pragma unroll
    for (int s = 0; s < 16; ++s) {
      union { uint4 q[2]; v16bf v; } u;
      u.q[0] = *(const uint4*)(pb + s * 64);
      u.q[1] = *(const uint4*)(pb + s * 64 + 16);
      Bf[s] = u.v;
    }
  }
  __syncthreads();

  float accWG = 0.f;

  for (int it = 0; it < BT_PER; ++it) {
    const int tile = wg + it * NWG;          // 0..1023
    const long rowBase = (long)tile * 16;

    __syncthreads();                          // LDS reuse guard

    // --- stage 16x512 X tile -> bf16 LDS; exact fp32 ||x||^2 per row
    if (tid < 512) {
      int m = tid >> 5;                       // row, whole wave per row
      int l = tid & 31;                       // 16 floats per lane
      const float* xr = x + (rowBase + m) * D_AMBI + l * 16;
      float v[16];
      float ss = 0.f;
#pragma unroll
      for (int i = 0; i < 4; ++i) {
        float4 f = ((const float4*)xr)[i];
        v[4 * i] = f.x; v[4 * i + 1] = f.y; v[4 * i + 2] = f.z; v[4 * i + 3] = f.w;
      }
#pragma unroll
      for (int i = 0; i < 16; ++i) ss = fmaf(v[i], v[i], ss);
#pragma unroll
      for (int msk = 16; msk >= 1; msk >>= 1) ss += __shfl_xor(ss, msk, 32);
      if (l == 0) xx[m] = ss;
      union { uint4 q[2]; __bf16 h[16]; } u;
#pragma unroll
      for (int i = 0; i < 16; ++i) u.h[i] = (__bf16)v[i];
      uint4* dst = (uint4*)(xt + m * XPITCH + l * 32);
      dst[0] = u.q[0];
      dst[1] = u.q[1];
    }
    __syncthreads();

    // --- 16 chained WMMAs over K=512 (A from LDS, B resident)
    v8f acc = {0.f, 0.f, 0.f, 0.f, 0.f, 0.f, 0.f, 0.f};
    {
      // A 16x32 bf16 layout: lane<16: K = e; lane>=16: K = 16 + e
      const int M = lane & 15;
      const int base = M * XPITCH + ((lane < 16) ? 0 : 32);
#pragma unroll
      for (int s = 0; s < 16; ++s) {
        union { uint4 q[2]; v16bf v; } u;
        u.q[0] = *(const uint4*)(xt + base + s * 64);        // K chunk lo
        u.q[1] = *(const uint4*)(xt + base + s * 64 + 16);   // K chunk hi (+8)
        acc = __builtin_amdgcn_wmma_f32_16x16x32_bf16(
            false, u.v, false, Bf[s], (short)0, acc, false, false);
      }
    }
    // --- D (16x16 f32): VGPR v holds M = v + (lane>=16 ? 8 : 0), N = lane%16
    {
      const int col  = w * 16 + (lane & 15);
      const int mofs = (lane < 16) ? 0 : 8;
#pragma unroll
      for (int v2 = 0; v2 < 8; ++v2)
        Yf[(v2 + mofs) * YPITCH + col] = acc[v2];
    }
    __syncthreads();

    // --- epilogue: obj[m,k] = xx - 2 w.z + z^T G z + LAMB*reg; min over k
    if (tid < 256) {
      const int m = tid >> 4, k = tid & 15;
      const float* zr = &Yf[m * YPITCH + k * 10];
      const float* wr = &Yf[m * YPITCH + 160 + k * 10];
      float z[10], wv[10];
#pragma unroll
      for (int i = 0; i < 10; ++i) { z[i] = zr[i]; wv[i] = wr[i]; }
      const float* g = &Gl[k * 100];
      float corr = 0.f;
#pragma unroll
      for (int i = 0; i < 10; ++i) {
        const float2* gp = (const float2*)(g + i * 10);
        float gi = 0.f;
#pragma unroll
        for (int j2 = 0; j2 < 5; ++j2) {
          float2 gg = gp[j2];
          gi = fmaf(gg.x, z[2 * j2], gi);
          gi = fmaf(gg.y, z[2 * j2 + 1], gi);
        }
        corr = fmaf(z[i], gi, corr);
        corr = fmaf(-2.f * z[i], wv[i], corr);
      }
      float obj = xx[m] + corr + LAMB_W * regl[k];
      obj = fminf(obj, __shfl_xor(obj, 1, 32));
      obj = fminf(obj, __shfl_xor(obj, 2, 32));
      obj = fminf(obj, __shfl_xor(obj, 4, 32));
      obj = fminf(obj, __shfl_xor(obj, 8, 32));
      if (k == 0) minv[m] = obj;
    }
    __syncthreads();
    if (tid == 0) {
      float s = 0.f;
#pragma unroll
      for (int m = 0; m < 16; ++m) s += minv[m];
      accWG += s;
    }
  }
  if (tid == 0) partials[wg] = accWG;
}

// --- deterministic fixed-order final reduction
__global__ void kmae_reduce(const float* __restrict__ partials,
                            float* __restrict__ out) {
  if (threadIdx.x == 0 && blockIdx.x == 0) {
    float s = 0.f;
    for (int i = 0; i < NWG; ++i) s += partials[i];
    out[0] = s / (float)BATCH;
  }
}

extern "C" void kernel_launch(void* const* d_in, const int* in_sizes, int n_in,
                              void* d_out, int out_size, void* d_ws, size_t ws_size,
                              hipStream_t stream) {
  const float* x  = (const float*)d_in[0];   // (16384, 512)
  const float* Us = (const float*)d_in[1];   // (16, 512, 10)
  const float* Vs = (const float*)d_in[2];   // (16, 10, 512)
  float* out = (float*)d_out;                // scalar f32

  unsigned char* ws = (unsigned char*)d_ws;
  __bf16* P    = (__bf16*)(ws + WS_P);
  float*  G    = (float*)(ws + WS_G);
  float*  reg  = (float*)(ws + WS_REG);
  float*  part = (float*)(ws + WS_PART);

  kmae_prep_P<<<NCOL, 256, 0, stream>>>(Us, Vs, P);
  kmae_prep_G<<<K_CL, 128, 0, stream>>>(Us, Vs, G, reg);
  kmae_main<<<NWG, WG_THREADS, 0, stream>>>(x, P, G, reg, part);
  kmae_reduce<<<1, 32, 0, stream>>>(part, out);
}
